// AttentionModuleLevel_73083163509265
// MI455X (gfx1250) — matrix operand order repaired and checked by reading, not compiled, not run on hardware
//
#include <hip/hip_runtime.h>

typedef __attribute__((ext_vector_type(16))) _Float16 v16h;
typedef __attribute__((ext_vector_type(8)))  float    v8f;

#define IMG_H 512
#define IMG_W 512
#define NBATCH 4
#define BK 11      // border kernel size
#define GK 23      // group kernel size
#define PADB 5
#define PADG 11
#define BP 80      // LDS row pitch (halves) for border input tiles (160B, 32B-aligned)
#define GP 80      // LDS row pitch (halves) for group input tiles

// Workspace layout (bytes):
//   [0, 33554432)                 : border map, f16, [4][16][512][512]
//   [33554432, 34308096)          : Atoep group-weight Toeplitz A-layout, f16, [16][23][2][32][16]
//   [34308096, 34398208)          : AbT  border-weight Toeplitz A-layout, f16, [8][11][32][16]
#define WS_BORDER_OFF 0
#define WS_ATOEP_OFF  33554432u
#define WS_ABT_OFF    (33554432u + 753664u)

// K index for 16-bit A-matrix element e (0..15) held by `lane` (contiguous, matches B):
//   lanes 0-15:  K 0..15
//   lanes 16-31: K 16..31
__device__ __forceinline__ int kmapA(int lane, int e) {
    return ((lane >= 16) ? 16 : 0) + e;
}

// ---------------------------------------------------------------------------
// Kernel 0: build Toeplitz weight layouts once (L2-resident thereafter).
// ---------------------------------------------------------------------------
__global__ __launch_bounds__(256) void prep_kernel(const float* __restrict__ Wb,
                                                   const float* __restrict__ Wg,
                                                   _Float16* __restrict__ Atoep,
                                                   _Float16* __restrict__ AbT) {
    int t = threadIdx.x;
    // Group weights as Toeplitz A: A(m,k) = Wg[c][dy][k-m], 0<=k-m<23.
    // Flat: (((c*23+dy)*2+q)*32 + lane)*16 + e
    for (int idx = t; idx < 16 * 23 * 2 * 32 * 16; idx += 256) {
        int e = idx & 15;
        int l = (idx >> 4) & 31;
        int q = (idx >> 9) & 1;
        int rest = idx >> 10;
        int dy = rest % 23;
        int c  = rest / 23;
        int m  = l & 15;
        int dx = kmapA(l, e) + 32 * q - m;
        float v = 0.f;
        if (dx >= 0 && dx < GK) v = Wg[(c * GK + dy) * GK + dx];
        Atoep[idx] = (_Float16)v;
    }
    // Border weights as Toeplitz A: A(m,k) = Wb[f][dy][k-m], 0<=k-m<11.
    // Span 16+10=26 <= 32 -> single K-chunk. Flat: ((f*11+dy)*32 + lane)*16 + e
    for (int idx = t; idx < 8 * 11 * 32 * 16; idx += 256) {
        int e = idx & 15;
        int l = (idx >> 4) & 31;
        int rest = idx >> 9;
        int dy = rest % 11;
        int f  = rest / 11;
        int m  = l & 15;
        int dx = kmapA(l, e) - m;
        float v = 0.f;
        if (dx >= 0 && dx < BK) v = Wb[(f * BK + dy) * BK + dx];
        AbT[idx] = (_Float16)v;
    }
}

// ---------------------------------------------------------------------------
// Kernel 1: border conv via Toeplitz WMMA + spikes + WTA logic -> border map.
// One block = one 16(y) x 64(x) tile (4 x-subtiles). 8 waves; wave w owns
// filter w; one A load per dy feeds 8 WMMAs (2 channels x 4 subtiles).
// ---------------------------------------------------------------------------
__global__ __launch_bounds__(256) void border_kernel(const float* __restrict__ inp,
                                                     const _Float16* __restrict__ AbT,
                                                     _Float16* __restrict__ borderBuf) {
    __shared__ __align__(32) _Float16 bT[2][26][BP];   // staged f16 input, halo 5, cols 74..79 zero
    __shared__ unsigned char posS[8][16][64];          // spike(conv ch0)
    __shared__ unsigned char negS[8][16][64];          // spike(conv ch1)

    const int b  = blockIdx.z;
    const int y0 = blockIdx.y * 16;
    const int x0 = blockIdx.x * 64;
    const int t    = threadIdx.x;
    const int lane = t & 31;
    const int wave = t >> 5;

    // Stage both input channels as f16 (zero-padded halo + col pad).
    for (int idx = t; idx < 2 * 26 * BP; idx += 256) {
        int col = idx % BP;
        int row = (idx / BP) % 26;
        int ch  = idx / (26 * BP);
        int gy = y0 - PADB + row, gx = x0 - PADB + col;
        float v = 0.f;
        if (col < 64 + 2 * PADB && gy >= 0 && gy < IMG_H && gx >= 0 && gx < IMG_W)
            v = inp[((size_t)(b * 2 + ch) * IMG_H + gy) * IMG_W + gx];
        bT[ch][row][col] = (_Float16)v;
    }
    __syncthreads();

    const int n    = lane & 15;   // D column = output y row ; also B column
    const int hsel = lane >> 4;   // which K half this lane holds

    // Wave w computes filter f=w; 2 channels x 4 x-subtiles share each A.
    const int f = wave;
    v8f acc[2][4];
    #pragma unroll
    for (int ch = 0; ch < 2; ++ch)
        #pragma unroll
        for (int s = 0; s < 4; ++s) acc[ch][s] = (v8f){};

    for (int dy = 0; dy < BK; ++dy) {
        v16h a = *(const v16h*)(AbT + ((size_t)(f * BK + dy) * 32 + lane) * 16);
        #pragma unroll
        for (int ch = 0; ch < 2; ++ch) {
            const _Float16* brow = &bT[ch][n + dy][0];
            #pragma unroll
            for (int s = 0; s < 4; ++s) {
                // B = contiguous input row segment, row n+dy per lane, 16 halves at col s*16+16*hsel.
                v16h bb = *(const v16h*)(brow + s * 16 + hsel * 16);
                acc[ch][s] = __builtin_amdgcn_wmma_f32_16x16x32_f16(false, a, false, bb,
                                                                    (short)0, acc[ch][s], false, false);
            }
        }
    }
    // D layout: column n = y, M(x within subtile) = j + 8*hsel.
    #pragma unroll
    for (int s = 0; s < 4; ++s)
        #pragma unroll
        for (int j = 0; j < 8; ++j) {
            int x = s * 16 + j + 8 * hsel;
            posS[f][n][x] = (acc[0][s][j] >= 1.0f) ? 1 : 0;
            negS[f][n][x] = (acc[1][s][j] >= 1.0f) ? 1 : 0;
        }
    __syncthreads();

    // Elementwise: each thread handles 4 pixels (one per x-subtile).
    const int y  = t >> 4;
    const int xl = t & 15;
    const int gy = y0 + y;
    const size_t chStride = (size_t)IMG_H * IMG_W;
    #pragma unroll
    for (int s = 0; s < 4; ++s) {
        const int x  = xl + 16 * s;
        const int gx = x0 + x;
        float vm = inp[((size_t)(b * 2 + 0) * IMG_H + gy) * IMG_W + gx] +
                   inp[((size_t)(b * 2 + 1) * IMG_H + gy) * IMG_W + gx];
        float b13[4], b24[4];
        float tmax = 0.f;
        #pragma unroll
        for (int o = 0; o < 4; ++o) {
            float pe = (float)posS[2 * o][y][x],     po = (float)posS[2 * o + 1][y][x];
            float ne = (float)negS[2 * o][y][x],     no = (float)negS[2 * o + 1][y][x];
            float s1 = (vm * (pe - 1.5f * no) >= 1.f) ? 1.f : 0.f;
            float s2 = (vm * (ne - 1.5f * po) >= 1.f) ? 1.f : 0.f;
            float s3 = (vm * (po - 1.5f * ne) >= 1.f) ? 1.f : 0.f;
            float s4 = (vm * (no - 1.5f * pe) >= 1.f) ? 1.f : 0.f;
            b13[o] = s1 + s2;
            b24[o] = s3 + s4;
            tmax = fmaxf(tmax, fabsf(b13[o] - b24[o]));
        }
        const size_t pixOff = (size_t)gy * IMG_W + gx;
        #pragma unroll
        for (int o = 0; o < 4; ++o) {
            float d   = b13[o] - b24[o];
            float wta = (fabsf(d) == tmax) ? 1.f : 0.f;
            float wd  = wta * d;
            float b1p = (wd  >= 1.f) ? 1.f : 0.f;
            float b1n = (-wd >= 1.f) ? 1.f : 0.f;
            size_t base = (size_t)(b * 16 + 4 * o) * chStride + pixOff;
            borderBuf[base]                = (_Float16)(b1p * b13[o]);
            borderBuf[base + chStride]     = (_Float16)(b1p * b24[o]);   // G_INH = 1.0
            borderBuf[base + 2 * chStride] = (_Float16)(b1n * b24[o]);
            borderBuf[base + 3 * chStride] = (_Float16)(b1n * b13[o]);   // G_INH = 1.0
        }
    }
}

// ---------------------------------------------------------------------------
// Kernel 2: depthwise 23x23 group conv via Toeplitz WMMA + spike combine.
// One block = one 16(y) x 32(x) tile (2 x-subtiles). 8 waves; channels in
// 2 passes of 8 (static LDS = 48.6KB tile + 8KB spikes < 64KB). Each (A,q)
// load feeds both x-subtiles.
// ---------------------------------------------------------------------------
__global__ __launch_bounds__(256) void group_kernel(const _Float16* __restrict__ borderBuf,
                                                    const _Float16* __restrict__ Atoep,
                                                    float* __restrict__ out) {
    __shared__ __align__(32) _Float16 bT[8][38][GP];  // 8 channels, halo 11, cols 54..79 zero
    __shared__ unsigned char gS[16][16][32];          // spike(group conv) for all 16 channels

    const int b  = blockIdx.z;
    const int y0 = blockIdx.y * 16;
    const int x0 = blockIdx.x * 32;
    const int t    = threadIdx.x;
    const int lane = t & 31;
    const int wave = t >> 5;
    const int n    = lane & 15;   // D column = output y row ; also B column
    const int hsel = lane >> 4;

    for (int pass = 0; pass < 2; ++pass) {
        const int cbase = pass * 8;
        // Stage 8 channels of the border map (zero pad cols >= 54 and halo OOB).
        for (int idx = t; idx < 8 * 38 * GP; idx += 256) {
            int col = idx % GP;
            int row = (idx / GP) % 38;
            int cc  = idx / (38 * GP);
            int gy = y0 - PADG + row, gx = x0 - PADG + col;
            _Float16 v = (_Float16)0.f;
            if (col < 32 + 2 * PADG && gy >= 0 && gy < IMG_H && gx >= 0 && gx < IMG_W)
                v = borderBuf[((size_t)(b * 16 + cbase + cc) * IMG_H + gy) * IMG_W + gx];
            bT[cc][row][col] = v;
        }
        __syncthreads();

        // One channel per wave; 2 x-subtiles share each Toeplitz A chunk.
        const int cc = wave;            // 0..7 within pass
        const int c  = cbase + cc;      // global channel
        v8f acc[2];
        acc[0] = (v8f){};
        acc[1] = (v8f){};
        for (int dy = 0; dy < GK; ++dy) {
            const _Float16* brow = &bT[cc][n + dy][0];
            v16h a0 = *(const v16h*)(Atoep + ((size_t)((c * GK + dy) * 2 + 0) * 32 + lane) * 16);
            v16h a1 = *(const v16h*)(Atoep + ((size_t)((c * GK + dy) * 2 + 1) * 32 + lane) * 16);
            #pragma unroll
            for (int s = 0; s < 2; ++s) {
                v16h b0 = *(const v16h*)(brow + s * 16 + 0 * 32 + hsel * 16);
                v16h b1 = *(const v16h*)(brow + s * 16 + 1 * 32 + hsel * 16);
                acc[s] = __builtin_amdgcn_wmma_f32_16x16x32_f16(false, a0, false, b0,
                                                                (short)0, acc[s], false, false);
                acc[s] = __builtin_amdgcn_wmma_f32_16x16x32_f16(false, a1, false, b1,
                                                                (short)0, acc[s], false, false);
            }
        }
        // D layout: column n = y, M(x within subtile) = j + 8*hsel.
        #pragma unroll
        for (int s = 0; s < 2; ++s)
            #pragma unroll
            for (int j = 0; j < 8; ++j) {
                int x = s * 16 + j + 8 * hsel;
                gS[c][n][x] = (acc[s][j] >= 1.f) ? 1 : 0;
            }
        __syncthreads();   // protects bT (WAR) before next pass's restage
    }

    // Combine: spike(g0-g1)+spike(g2-g3) summed over orientations; g in {0,1}.
    const int y  = t >> 4;
    const int xl = t & 15;
    #pragma unroll
    for (int s = 0; s < 2; ++s) {
        const int x = xl + 16 * s;
        float o = 0.f;
        #pragma unroll
        for (int ori = 0; ori < 4; ++ori) {
            int g0 = gS[4 * ori + 0][y][x];
            int g1 = gS[4 * ori + 1][y][x];
            int g2 = gS[4 * ori + 2][y][x];
            int g3 = gS[4 * ori + 3][y][x];
            o += (float)((g0 & ~g1) + (g2 & ~g3));
        }
        out[((size_t)b * IMG_H + (y0 + y)) * IMG_W + (x0 + x)] = o;
    }
}

// ---------------------------------------------------------------------------
extern "C" void kernel_launch(void* const* d_in, const int* in_sizes, int n_in,
                              void* d_out, int out_size, void* d_ws, size_t ws_size,
                              hipStream_t stream) {
    const float* inp = (const float*)d_in[0];   // (4,2,512,512) f32
    const float* Wb  = (const float*)d_in[1];   // (8,1,11,11)  f32
    const float* Wg  = (const float*)d_in[2];   // (16,1,23,23) f32

    _Float16* borderBuf = (_Float16*)((char*)d_ws + WS_BORDER_OFF);
    _Float16* Atoep     = (_Float16*)((char*)d_ws + WS_ATOEP_OFF);
    _Float16* AbT       = (_Float16*)((char*)d_ws + WS_ABT_OFF);
    float* out = (float*)d_out;

    prep_kernel<<<1, 256, 0, stream>>>(Wb, Wg, Atoep, AbT);

    dim3 gridB(IMG_W / 64, IMG_H / 16, NBATCH);   // 8 x 32 x 4 tiles
    border_kernel<<<gridB, 256, 0, stream>>>(inp, AbT, borderBuf);

    dim3 gridG(IMG_W / 32, IMG_H / 16, NBATCH);   // 16 x 32 x 4 tiles
    group_kernel<<<gridG, 256, 0, stream>>>(borderBuf, Atoep, out);
}
